// paraCLSTM_4389456576650
// MI455X (gfx1250) — matrix operand order repaired and checked by reading, not compiled, not run on hardware
//
#include <hip/hip_runtime.h>
#include <math.h>

// ---- problem constants ----
#define BB 64
#define TT 512
#define TP2 (TT+2)   // padded time extent (zero row each side)
#define EE 256
#define UU 256
#define GG 1024      // 4*U
#define OO 20
#define TPOOL 128    // T/4
#define FF (TPOOL*256) // 32768

typedef __attribute__((ext_vector_type(8)))  float  v8f;
typedef __attribute__((ext_vector_type(16))) __bf16 v16bf;
typedef __attribute__((ext_vector_type(8)))  __bf16 v8bf;

static __device__ __forceinline__ v8f zero8() {
    v8f z = {0.f,0.f,0.f,0.f,0.f,0.f,0.f,0.f};
    return z;
}

// A fragment (16x32 bf16, row-major source). `row` = this lane's row start.
// Layout: lanes 0-15: elems0-15 <-> K0-15
//         lanes16-31: elems0-15 <-> K16-31
static __device__ __forceinline__ v16bf load_a_frag(const __bf16* row, int k0, int lane) {
    int kb = ((lane >> 4) << 4);           // 0 or 16
    const __bf16* p = row + k0 + kb;
    union { v16bf v; v8bf h[2]; } u;
    u.h[0] = *(const v8bf*)(p);
    u.h[1] = *(const v8bf*)(p + 8);
    return u.v;
}

static __device__ __forceinline__ v8f wmma_bf16(v16bf a, v16bf b, v8f c) {
    return __builtin_amdgcn_wmma_f32_16x16x32_bf16(false, a, false, b, (short)0, c, false, false);
}
static __device__ __forceinline__ float sigm(float x) {
    return 1.f / (1.f + __expf(-x));
}

// ---------------- K0: generic f32 [R,C] -> bf16 transposed [C,R] ----------------
__global__ void k_transpose_bf16(const float* __restrict__ src, __bf16* __restrict__ dst,
                                 int R, int C) {
    long i = (long)blockIdx.x * 256 + threadIdx.x;
    if (i >= (long)R * C) return;
    int r = (int)(i / C), c = (int)(i % C);
    dst[(size_t)c * R + r] = (__bf16)src[i];
}

// ---------------- K1a: embedding gather -> bf16 padded [B, T+2, E] ----------------
__global__ void k_embed(const int* __restrict__ tokens, const float* __restrict__ emb,
                        __bf16* __restrict__ e) {
    int bt = blockIdx.x;               // 0..B*T-1
    int i  = threadIdx.x;
    int b  = bt >> 9, t = bt & 511;
    int tok = tokens[bt];
    e[((size_t)b * TP2 + 1 + t) * EE + i] = (__bf16)emb[(size_t)tok * EE + i];
}

// K1b: zero the 2 pad rows per batch (SAME-padding halo)
__global__ void k_zero_pad(__bf16* __restrict__ e) {
    int b = blockIdx.x >> 1;
    int side = blockIdx.x & 1;         // 0: row -1, 1: row T
    size_t row = (size_t)b * TP2 + (size_t)side * (TT + 1);
    e[row * EE + threadIdx.x] = (__bf16)0.f;
}

// ---------------- K2: conv1d(same) + relu + maxpool4, WMMA bf16 ----------------
// grid (256,16), block 256 (8 waves). wave tile: 16 t-rows x 16 out-channels.
// Branch-free: halo rows come from the zero-padded e layout.
__global__ __launch_bounds__(256)
void k_conv_pool(const __bf16* __restrict__ e, const __bf16* __restrict__ convT,
                 const float* __restrict__ conv_b, __bf16* __restrict__ f) {
    int wv = threadIdx.x >> 5, lane = threadIdx.x & 31;
    int mt = blockIdx.x * 8 + wv;      // 0..2047
    int nt = blockIdx.y;               // 0..15
    int m0 = mt * 16, n0 = nt * 16;
    int b  = m0 >> 9;
    int t0 = m0 & 511;

    v8f acc = zero8();
    int tlane = t0 + (lane & 15);      // this lane's t within batch
    // padded row index for tap w: b*TP2 + 1 + tlane + (w-1) -> base at w=0 is tlane
    const __bf16* arow0 = e + ((size_t)b * TP2 + tlane) * EE;
#pragma unroll
    for (int w = 0; w < 3; ++w) {
        const __bf16* arow  = arow0 + (size_t)w * EE;
        const __bf16* bbase = convT + ((size_t)w * 256 + n0 + (lane & 15)) * EE + ((lane >> 4) << 4);
#pragma unroll
        for (int k0 = 0; k0 < EE; k0 += 32) {
            v16bf a  = load_a_frag(arow, k0, lane);
            v16bf bf = *(const v16bf*)(bbase + k0);
            acc = wmma_bf16(a, bf, acc);
        }
    }
    // bias + relu + pool (rows 0-3 -> acc0-3 lo lanes, 4-7 -> acc4-7 lo,
    //                     8-11 -> acc0-3 hi lanes, 12-15 -> acc4-7 hi)
    int n = n0 + (lane & 15);
    float bias = conv_b[n];
    float v0 = fmaxf(fmaxf(acc[0], acc[1]), fmaxf(acc[2], acc[3]));
    float v1 = fmaxf(fmaxf(acc[4], acc[5]), fmaxf(acc[6], acc[7]));
    v0 = fmaxf(v0 + bias, 0.f);
    v1 = fmaxf(v1 + bias, 0.f);
    int pr = (t0 >> 2) + ((lane >> 4) << 1);
    f[(size_t)b * FF + (size_t)pr * 256 + n]       = (__bf16)v0;
    f[(size_t)b * FF + (size_t)(pr + 1) * 256 + n] = (__bf16)v1;
}

// ---------------- K3: zx = e @ wxT + b, t-major bf16 output [T][B][G] ----------------
// grid (256,16,2), block 256. wave tile: 16 rows x 64 cols (4 accumulators).
__global__ __launch_bounds__(256)
void k_zx(const __bf16* __restrict__ e,
          const __bf16* __restrict__ wxT_f, const __bf16* __restrict__ wxT_b,
          const float* __restrict__ b_f, const float* __restrict__ b_b,
          __bf16* __restrict__ zx_f, __bf16* __restrict__ zx_b) {
    int wv = threadIdx.x >> 5, lane = threadIdx.x & 31;
    int mt = blockIdx.x * 8 + wv;   // 0..2047
    int ng = blockIdx.y;            // 0..15 (groups of 64 cols)
    int dir = blockIdx.z;
    const __bf16* wx = dir ? wxT_b : wxT_f;
    const float*  bs = dir ? b_b   : b_f;
    __bf16*       zx = dir ? zx_b  : zx_f;

    int m0 = mt * 16;
    int b  = m0 >> 9;
    int t0 = m0 & 511;
    int tl = t0 + (lane & 15);
    int tsrc = dir ? (TT - 1 - tl) : tl;
    const __bf16* arow = e + ((size_t)b * TP2 + 1 + tsrc) * EE;

    v8f acc[4] = {zero8(), zero8(), zero8(), zero8()};
#pragma unroll
    for (int k0 = 0; k0 < EE; k0 += 32) {
        v16bf a = load_a_frag(arow, k0, lane);
#pragma unroll
        for (int j = 0; j < 4; ++j) {
            int n0 = ng * 64 + j * 16;
            const __bf16* bbase = wx + (size_t)(n0 + (lane & 15)) * EE + ((lane >> 4) << 4);
            v16bf bf = *(const v16bf*)(bbase + k0);
            acc[j] = wmma_bf16(a, bf, acc[j]);
        }
    }
    int mb = (lane >> 4) << 3;
#pragma unroll
    for (int j = 0; j < 4; ++j) {
        int n = ng * 64 + j * 16 + (lane & 15);
        float bias = bs[n];
#pragma unroll
        for (int r = 0; r < 8; ++r) {
            int m  = m0 + mb + r;
            int ti = m & 511;
            int bi = m >> 9;
            zx[((size_t)ti * BB + bi) * GG + n] = (__bf16)(acc[j][r] + bias);
        }
    }
}

// ---------------- K4: LSTM recurrence, 1 workgroup per direction ----------------
// block 1024 (32 waves). LDS: h bf16 32KB + z bf16 128KB = 160KB.
__global__ __launch_bounds__(1024)
void k_lstm(const __bf16* __restrict__ zx_f, const __bf16* __restrict__ zx_b,
            const __bf16* __restrict__ whT_f, const __bf16* __restrict__ whT_b,
            float* __restrict__ c_ws, __bf16* __restrict__ r_out) {
    __shared__ __bf16 h_lds[BB * UU];   // 32 KB
    __shared__ __bf16 z_lds[BB * GG];   // 128 KB
    int dir = blockIdx.x;
    const __bf16* zx = dir ? zx_b : zx_f;
    const __bf16* wh = dir ? whT_b : whT_f;
    float* cst = c_ws + (size_t)dir * BB * UU;
    int tid = threadIdx.x;

    for (int i = tid; i < BB * UU; i += 1024) { h_lds[i] = (__bf16)0.f; cst[i] = 0.f; }
    __syncthreads();

    int wv = tid >> 5, lane = tid & 31;
    int mb = (lane >> 4) << 3;

    for (int t = 0; t < TT; ++t) {
        const __bf16* zrow = zx + (size_t)t * BB * GG;
        // z = zx_t + h @ whT : wave wv owns N-tiles 2wv..2wv+1, all 4 M-tiles
#pragma unroll
        for (int j = 0; j < 2; ++j) {
            int n0 = (wv * 2 + j) * 16;
            int n  = n0 + (lane & 15);
            const __bf16* bbase = wh + (size_t)n * UU + ((lane >> 4) << 4);
#pragma unroll
            for (int mtile = 0; mtile < 4; ++mtile) {
                int m0 = mtile * 16;
                v8f acc;
#pragma unroll
                for (int r = 0; r < 8; ++r)
                    acc[r] = (float)zrow[(size_t)(m0 + mb + r) * GG + n];
                const __bf16* arow = h_lds + (size_t)(m0 + (lane & 15)) * UU;
#pragma unroll
                for (int k0 = 0; k0 < UU; k0 += 32) {
                    v16bf a  = load_a_frag(arow, k0, lane);
                    v16bf bf = *(const v16bf*)(bbase + k0);
                    acc = wmma_bf16(a, bf, acc);
                }
#pragma unroll
                for (int r = 0; r < 8; ++r)
                    z_lds[(size_t)(m0 + mb + r) * GG + n] = (__bf16)acc[r];
            }
        }
        __syncthreads();
        // gates: thread -> (b = tid/16, 16 consecutive u)
        {
            int bi = tid >> 4;
            int u0 = (tid & 15) << 4;
            const __bf16* zr = z_lds + (size_t)bi * GG;
#pragma unroll
            for (int u = u0; u < u0 + 16; ++u) {
                float zi = (float)zr[u];
                float zf = (float)zr[256 + u];
                float zg = (float)zr[512 + u];
                float zo = (float)zr[768 + u];
                float c  = sigm(zf) * cst[bi * UU + u] + sigm(zi) * tanhf(zg);
                cst[bi * UU + u]  = c;
                h_lds[bi * UU + u] = (__bf16)(sigm(zo) * tanhf(c));
            }
        }
        __syncthreads();
    }
    // write final h into r = concat(hf, hb) bf16 [B, 512]
    for (int i = tid; i < BB * UU; i += 1024) {
        int bi = i >> 8, u = i & 255;
        r_out[(size_t)bi * 512 + dir * 256 + u] = h_lds[i];
    }
}

// ---------------- K5: d1 split-K partials (deterministic) ----------------
// grid 128, block 256. waveid -> (mt, nt, kc); K chunk = 2048.
__global__ __launch_bounds__(256)
void k_d1_partial(const __bf16* __restrict__ f, const __bf16* __restrict__ d1wT,
                  float* __restrict__ part) {
    int wv = threadIdx.x >> 5, lane = threadIdx.x & 31;
    int id = blockIdx.x * 8 + wv;       // 0..1023
    int kc = id & 15;
    int nt = (id >> 4) & 15;
    int mt = id >> 8;                   // 0..3
    const __bf16* arow  = f + (size_t)(mt * 16 + (lane & 15)) * FF;
    const __bf16* bbase = d1wT + (size_t)(nt * 16 + (lane & 15)) * FF + ((lane >> 4) << 4);
    v8f acc = zero8();
    int kbeg = kc * 2048;
#pragma unroll 4
    for (int k0 = kbeg; k0 < kbeg + 2048; k0 += 32) {
        v16bf a  = load_a_frag(arow, k0, lane);
        v16bf bf = *(const v16bf*)(bbase + k0);
        acc = wmma_bf16(a, bf, acc);
    }
    int mb = (lane >> 4) << 3;
    int n  = nt * 16 + (lane & 15);
#pragma unroll
    for (int r = 0; r < 8; ++r)
        part[(size_t)kc * (BB * 256) + (size_t)(mt * 16 + mb + r) * 256 + n] = acc[r];
}

// grid 64, block 256: con[:, 0:256] = sum_kc part + d1_b
__global__ void k_d1_reduce(const float* __restrict__ part, const float* __restrict__ d1_b,
                            float* __restrict__ con) {
    int b = blockIdx.x, n = threadIdx.x;
    float s = d1_b[n];
#pragma unroll
    for (int kc = 0; kc < 16; ++kc)
        s += part[(size_t)kc * (BB * 256) + (size_t)b * 256 + n];
    con[(size_t)b * 512 + n] = s;
}

// ---------------- K6: d2 = r @ d2wT + d2_b -> con[:, 256:512] ----------------
// grid 8, block 256 (64 wave tiles), K=512.
__global__ __launch_bounds__(256)
void k_d2(const __bf16* __restrict__ r, const __bf16* __restrict__ d2wT,
          const float* __restrict__ d2_b, float* __restrict__ con) {
    int wv = threadIdx.x >> 5, lane = threadIdx.x & 31;
    int id = blockIdx.x * 8 + wv;   // 0..63
    int mt = id >> 4;               // 0..3
    int nt = id & 15;               // 0..15
    const __bf16* arow  = r + (size_t)(mt * 16 + (lane & 15)) * 512;
    const __bf16* bbase = d2wT + (size_t)(nt * 16 + (lane & 15)) * 512 + ((lane >> 4) << 4);
    v8f acc = zero8();
#pragma unroll
    for (int k0 = 0; k0 < 512; k0 += 32) {
        v16bf a  = load_a_frag(arow, k0, lane);
        v16bf bf = *(const v16bf*)(bbase + k0);
        acc = wmma_bf16(a, bf, acc);
    }
    int mb = (lane >> 4) << 3;
    int n  = nt * 16 + (lane & 15);
    float bias = d2_b[n];
#pragma unroll
    for (int r8 = 0; r8 < 8; ++r8)
        con[(size_t)(mt * 16 + mb + r8) * 512 + 256 + n] = acc[r8] + bias;
}

// ---------------- K7: out = softmax(con @ out_w + out_b) ----------------
// grid 64, block 32 (one wave per batch row).
__global__ void k_out(const float* __restrict__ con, const float* __restrict__ out_w,
                      const float* __restrict__ out_b, float* __restrict__ out) {
    int b = blockIdx.x;
    int o = threadIdx.x;
    bool valid = (o < OO);
    float acc = valid ? out_b[o] : -1e30f;
    if (valid) {
        const float* cr = con + (size_t)b * 512;
        for (int k = 0; k < 512; ++k)
            acc += cr[k] * out_w[(size_t)k * OO + o];
    }
    float m = acc;
    for (int off = 16; off; off >>= 1) m = fmaxf(m, __shfl_xor(m, off, 32));
    float e = valid ? __expf(acc - m) : 0.f;
    float s = e;
    for (int off = 16; off; off >>= 1) s += __shfl_xor(s, off, 32);
    if (valid) out[(size_t)b * OO + o] = e / s;
}

// ---------------- launch ----------------
extern "C" void kernel_launch(void* const* d_in, const int* in_sizes, int n_in,
                              void* d_out, int out_size, void* d_ws, size_t ws_size,
                              hipStream_t stream) {
    const int*   tokens = (const int*)  d_in[0];
    const float* emb    = (const float*)d_in[1];
    const float* conv_w = (const float*)d_in[2];
    const float* conv_b = (const float*)d_in[3];
    const float* d1_w   = (const float*)d_in[4];
    const float* d1_b   = (const float*)d_in[5];
    const float* wx_f   = (const float*)d_in[6];
    const float* wh_f   = (const float*)d_in[7];
    const float* b_f    = (const float*)d_in[8];
    const float* wx_b   = (const float*)d_in[9];
    const float* wh_b   = (const float*)d_in[10];
    const float* b_b    = (const float*)d_in[11];
    const float* d2_w   = (const float*)d_in[12];
    const float* d2_b   = (const float*)d_in[13];
    const float* out_w  = (const float*)d_in[14];
    const float* out_b  = (const float*)d_in[15];
    float* out = (float*)d_out;

    char* w = (char*)d_ws;
    size_t off = 0;
    auto alloc = [&](size_t bytes) {
        size_t o = off;
        off = (off + bytes + 255) & ~(size_t)255;
        return o;
    };
    __bf16* e_bf   = (__bf16*)(w + alloc((size_t)BB*TP2*EE*2)); // padded
    __bf16* wxT_f  = (__bf16*)(w + alloc((size_t)GG*EE*2));
    __bf16* whT_f  = (__bf16*)(w + alloc((size_t)GG*UU*2));
    __bf16* wxT_b  = (__bf16*)(w + alloc((size_t)GG*EE*2));
    __bf16* whT_b  = (__bf16*)(w + alloc((size_t)GG*UU*2));
    __bf16* convT  = (__bf16*)(w + alloc((size_t)3*256*EE*2));
    __bf16* d1wT   = (__bf16*)(w + alloc((size_t)256*FF*2));
    __bf16* d2wT   = (__bf16*)(w + alloc((size_t)256*512*2));
    __bf16* zx_f   = (__bf16*)(w + alloc((size_t)TT*BB*GG*2));
    __bf16* zx_b   = (__bf16*)(w + alloc((size_t)TT*BB*GG*2));
    __bf16* f_bf   = (__bf16*)(w + alloc((size_t)BB*FF*2));
    __bf16* r_bf   = (__bf16*)(w + alloc((size_t)BB*512*2));
    float*  c_ws   = (float*) (w + alloc((size_t)2*BB*UU*4));
    float*  part   = (float*) (w + alloc((size_t)16*BB*256*4));
    float*  con    = (float*) (w + alloc((size_t)BB*512*4));
    (void)ws_size; (void)in_sizes; (void)n_in; (void)out_size;

    // weight preprocessing (bf16 + transpose to [N,K])
    auto tgrid = [](long n) { return (unsigned)((n + 255) / 256); };
    k_transpose_bf16<<<tgrid((long)EE*GG), 256, 0, stream>>>(wx_f, wxT_f, EE, GG);
    k_transpose_bf16<<<tgrid((long)UU*GG), 256, 0, stream>>>(wh_f, whT_f, UU, GG);
    k_transpose_bf16<<<tgrid((long)EE*GG), 256, 0, stream>>>(wx_b, wxT_b, EE, GG);
    k_transpose_bf16<<<tgrid((long)UU*GG), 256, 0, stream>>>(wh_b, whT_b, UU, GG);
    for (int wtap = 0; wtap < 3; ++wtap)
        k_transpose_bf16<<<tgrid((long)EE*256), 256, 0, stream>>>(
            conv_w + (size_t)wtap*EE*256, convT + (size_t)wtap*256*EE, EE, 256);
    k_transpose_bf16<<<tgrid((long)FF*256), 256, 0, stream>>>(d1_w, d1wT, FF, 256);
    k_transpose_bf16<<<tgrid((long)512*256), 256, 0, stream>>>(d2_w, d2wT, 512, 256);

    // pipeline
    k_embed<<<BB*TT, EE, 0, stream>>>(tokens, emb, e_bf);
    k_zero_pad<<<BB*2, EE, 0, stream>>>(e_bf);
    k_conv_pool<<<dim3(256, 16), 256, 0, stream>>>(e_bf, convT, conv_b, f_bf);
    k_zx<<<dim3(256, 16, 2), 256, 0, stream>>>(e_bf, wxT_f, wxT_b, b_f, b_b, zx_f, zx_b);
    k_lstm<<<2, 1024, 0, stream>>>(zx_f, zx_b, whT_f, whT_b, c_ws, r_bf);
    k_d1_partial<<<128, 256, 0, stream>>>(f_bf, d1wT, part);
    k_d1_reduce<<<64, 256, 0, stream>>>(part, d1_b, con);
    k_d2<<<8, 256, 0, stream>>>(r_bf, d2wT, d2_b, con);
    k_out<<<64, 32, 0, stream>>>(con, out_w, out_b, out);
}
